// SCDM_TP_19387482374387
// MI455X (gfx1250) — matrix operand order repaired and checked by reading, not compiled, not run on hardware
//
#include <hip/hip_runtime.h>

#define N_NODES 100000
#define N_EDGES 800000
#define EMB 16
#define HID 128
#define OUTD 3
#define NTILES 6250            // N_NODES / 16, exact
#define WAVES_PER_BLK 8
#define NEG_SLOPE 0.01f

typedef __attribute__((ext_vector_type(16))) __bf16 v16bf;
typedef __attribute__((ext_vector_type(8)))  float  v8f;
typedef __attribute__((ext_vector_type(4)))  unsigned int u32x4;
typedef __attribute__((ext_vector_type(8)))  int i32x8;
typedef __attribute__((ext_vector_type(4)))  int i32x4;

__device__ __forceinline__ unsigned short f2bf_bits(float f) {
    __bf16 h = (__bf16)f;
    return __builtin_bit_cast(unsigned short, h);
}

// TDM: DMA a contiguous 128x128 bf16 (32KB) tile from global into LDS.
// D# built per CDNA5 ISA ch.8: group0 = {count=1, lds_addr, global_addr, type=2},
// group1 = {data_size=2B, tensor_dim0/1=128, tile_dim0/1=128, dim0_stride=128}.
// Toolchain here uses the 6-arg builtin: (u32x4, i32x8, i32x4, i32x4, i32x8, i32 cpol).
__device__ __forceinline__ void tdm_load_w128x128_bf16(const unsigned short* gsrc,
                                                       unsigned lds_off) {
    unsigned long long ga = (unsigned long long)gsrc;
    u32x4 g0;
    g0[0] = 1u;                                          // count=1, user descriptor
    g0[1] = lds_off;                                     // lds_addr (bytes)
    g0[2] = (unsigned)(ga & 0xFFFFFFFFu);                // global_addr[31:0]
    g0[3] = (unsigned)((ga >> 32) & 0x01FFFFFFu)         // global_addr[56:32]
          | (2u << 30);                                  // type=2 ("image")
    i32x8 g1;
    g1[0] = (1 << 16);                                   // data_size=1 (2 bytes), mask=0
    g1[1] = (HID << 16);                                 // tensor_dim0 = 128 (bits 79:48)
    g1[2] = (HID << 16);                                 // tensor_dim1 = 128 (bits 111:80)
    g1[3] = (HID << 16);                                 // tile_dim0  = 128 (bits 127:112)
    g1[4] = HID;                                         // tile_dim1  = 128, tile_dim2 = 0
    g1[5] = HID;                                         // tensor_dim0_stride = 128
    g1[6] = 0;
    g1[7] = 0;
    i32x4 gz4 = {0, 0, 0, 0};                            // groups 2/3 unused (2D tensor)
    i32x8 gz8 = {0, 0, 0, 0, 0, 0, 0, 0};
    __builtin_amdgcn_tensor_load_to_lds(g0, g1, gz4, gz4, gz8, 0);
}

// ---------------------------------------------------------------- prep: f32 weights -> bf16
// wIn is zero-padded to 32x128 so the MLP B-tile load needs no predication.
__global__ __launch_bounds__(256) void scdm_prep_weights(
    const float* __restrict__ W_in, const float* __restrict__ W_root,
    const float* __restrict__ W_rel,
    unsigned short* __restrict__ wIn, unsigned short* __restrict__ wRoot,
    unsigned short* __restrict__ wRel) {
    int t = blockIdx.x * 256 + threadIdx.x;          // grid covers 32768
    if (t < 32 * HID)        wIn[t]   = (t < EMB * HID) ? f2bf_bits(W_in[t]) : (unsigned short)0;
    if (t < HID * HID)       wRoot[t] = f2bf_bits(W_root[t]);
    if (t < 2 * HID * HID)   wRel[t]  = f2bf_bits(W_rel[t]);
}

// ---------------------------------------------------------------- zero scratch
__global__ __launch_bounds__(256) void scdm_zero(float* __restrict__ p, long long n) {
    long long i = (long long)blockIdx.x * blockDim.x + threadIdx.x;
    long long stride = (long long)gridDim.x * blockDim.x;
    for (; i < n; i += stride) p[i] = 0.0f;
}

// ---------------------------------------------------------------- per-relation in-degree
__global__ __launch_bounds__(256) void scdm_count(
    const int* __restrict__ index, const int* __restrict__ etype,
    float* __restrict__ cnt) {
    int e = blockIdx.x * 256 + threadIdx.x;          // 800000 = 3125*256, exact
    int dst = index[N_EDGES + e];
    int r = etype[e];
    atomicAdd(&cnt[(long long)r * N_NODES + dst], 1.0f);
}

// ---------------------------------------------------------------- edge scatter-add (f32 atomics, L2 resident)
__global__ __launch_bounds__(256) void scdm_scatter(
    const int* __restrict__ index, const int* __restrict__ etype,
    const float* __restrict__ x, float* __restrict__ agg) {
    long long t = (long long)blockIdx.x * 256 + threadIdx.x; // N_EDGES*32 threads
    int e = (int)(t >> 5);
    int c = ((int)t & 31) * 4;
    int src = index[e];
    int dst = index[N_EDGES + e];
    int r   = etype[e];
    const float4 xs = *(const float4*)(x + (long long)src * HID + c);
    float* ag = agg + ((long long)r * N_NODES + dst) * HID + c;
    atomicAdd(ag + 0, xs.x);
    atomicAdd(ag + 1, xs.y);
    atomicAdd(ag + 2, xs.z);
    atomicAdd(ag + 3, xs.w);
}

// ---------------------------------------------------------------- input MLP: leakyrelu(feat @ W_in + b_in), WMMA
__global__ __launch_bounds__(256) void scdm_mlp_in(
    const float* __restrict__ feat, const unsigned short* __restrict__ wIn,
    const float* __restrict__ b_in, float* __restrict__ xout) {
    int lane = threadIdx.x & 31;
    int wave = threadIdx.x >> 5;
    long long tile = (long long)blockIdx.x * WAVES_PER_BLK + wave;
    if (tile >= NTILES) return;                       // wave-uniform
    int m = lane & 15, half = lane >> 4;
    long long node = tile * 16 + m;

    // A tile: 16 nodes x K(=32, K>=16 zero-padded; no predicated loads).
    const float4* fp = (const float4*)(feat + node * EMB + half * 8);
    float4 f0 = fp[0];
    float4 f1 = fp[1];
    v16bf a;
    a[0] = (__bf16)f0.x;  a[1] = (__bf16)f0.y;  a[2] = (__bf16)f0.z;  a[3] = (__bf16)f0.w;
    a[4] = (__bf16)f1.x;  a[5] = (__bf16)f1.y;  a[6] = (__bf16)f1.z;  a[7] = (__bf16)f1.w;
#pragma unroll
    for (int e = 8; e < 16; ++e) a[e] = (__bf16)0.0f;

    // Preload all 8 B-tiles (column n = lane%16, element s -> k in table order; rows 16..31 are zero padding).
    v16bf bt[8];
#pragma unroll
    for (int nt = 0; nt < 8; ++nt)
#pragma unroll
        for (int s = 0; s < 16; ++s) {
            int k = (s < 8) ? (half * 8 + s) : (16 + half * 8 + (s - 8));
            bt[nt][s] = __builtin_bit_cast(__bf16, wIn[k * HID + nt * 16 + m]);
        }

#pragma unroll
    for (int nt = 0; nt < 8; ++nt) {
        float bv = b_in[nt * 16 + m];
        v8f c;
#pragma unroll
        for (int i = 0; i < 8; ++i) c[i] = bv;
        c = __builtin_amdgcn_wmma_f32_16x16x32_bf16(false, a, false, bt[nt], (short)0, c, false, false);
#pragma unroll
        for (int i = 0; i < 8; ++i) {
            float v = c[i];
            v = (v > 0.0f) ? v : NEG_SLOPE * v;       // leaky relu fused
            long long row = tile * 16 + i + 8 * half;
            xout[row * HID + nt * 16 + m] = v;
        }
    }
}

// ---------------------------------------------------------------- fused RGCN node update:
// out = x@W_root + b + (agg0/max(cnt0,1))@W_rel0 + (agg1/max(cnt1,1))@W_rel1
__global__ __launch_bounds__(256) void scdm_rgcn_node(
    const float* __restrict__ x, const float* __restrict__ agg,
    const float* __restrict__ cnt,
    const unsigned short* __restrict__ wRoot, const unsigned short* __restrict__ wRel,
    const float* __restrict__ bias, float* __restrict__ out) {
    __shared__ __align__(16) unsigned short sW[HID * HID];   // 32 KB, one weight matrix at a time

    int lane = threadIdx.x & 31;
    int wave = threadIdx.x >> 5;
    long long tile = (long long)blockIdx.x * WAVES_PER_BLK + wave;
    bool active = (tile < NTILES);                    // wave-uniform
    int m = lane & 15, half = lane >> 4;
    long long node = tile * 16 + m;

    v8f acc[8];
#pragma unroll
    for (int nt = 0; nt < 8; ++nt) {
        float bv = active ? bias[nt * 16 + m] : 0.0f;
#pragma unroll
        for (int i = 0; i < 8; ++i) acc[nt][i] = bv;
    }

    for (int sel = 0; sel < 3; ++sel) {
        const unsigned short* Wg = (sel == 0) ? wRoot : (wRel + (long long)(sel - 1) * HID * HID);
        __syncthreads();
        if (wave == 0) {                               // one TDM per block per phase
            tdm_load_w128x128_bf16(Wg, (unsigned)(unsigned long long)(void*)&sW[0]);
            __builtin_amdgcn_s_wait_tensorcnt(0);      // TENSORcnt -> 0
        }
        __syncthreads();

        if (active) {
            const float* src;
            float scale = 1.0f;
            if (sel == 0) {
                src = x + node * HID;
            } else {
                src = agg + ((long long)(sel - 1) * N_NODES + node) * HID;
                float c = cnt[(long long)(sel - 1) * N_NODES + node];
                scale = 1.0f / fmaxf(c, 1.0f);
            }
            const float4* sp = (const float4*)src;     // rows 512B aligned
#pragma unroll
            for (int kt = 0; kt < 4; ++kt) {
                // A element K map: e=0..7 -> kt*32 + half*8 + e ; e=8..15 -> +16
                float4 r0 = sp[kt * 8 + half * 2 + 0];
                float4 r1 = sp[kt * 8 + half * 2 + 1];
                float4 r2 = sp[kt * 8 + 4 + half * 2 + 0];
                float4 r3 = sp[kt * 8 + 4 + half * 2 + 1];
                v16bf a;
                a[0]  = (__bf16)(r0.x * scale); a[1]  = (__bf16)(r0.y * scale);
                a[2]  = (__bf16)(r0.z * scale); a[3]  = (__bf16)(r0.w * scale);
                a[4]  = (__bf16)(r1.x * scale); a[5]  = (__bf16)(r1.y * scale);
                a[6]  = (__bf16)(r1.z * scale); a[7]  = (__bf16)(r1.w * scale);
                a[8]  = (__bf16)(r2.x * scale); a[9]  = (__bf16)(r2.y * scale);
                a[10] = (__bf16)(r2.z * scale); a[11] = (__bf16)(r2.w * scale);
                a[12] = (__bf16)(r3.x * scale); a[13] = (__bf16)(r3.y * scale);
                a[14] = (__bf16)(r3.z * scale); a[15] = (__bf16)(r3.w * scale);

                v16bf bt[8];
#pragma unroll
                for (int nt = 0; nt < 8; ++nt)
#pragma unroll
                    for (int s = 0; s < 16; ++s) {
                        int krow = kt * 32 + ((s < 8) ? (half * 8 + s) : (16 + half * 8 + (s - 8)));
                        bt[nt][s] = __builtin_bit_cast(__bf16, sW[krow * HID + nt * 16 + m]);
                    }
#pragma unroll
                for (int nt = 0; nt < 8; ++nt) {
                    acc[nt] = __builtin_amdgcn_wmma_f32_16x16x32_bf16(
                        false, a, false, bt[nt], (short)0, acc[nt], false, false);
                }
            }
        }
    }

    if (active) {
#pragma unroll
        for (int nt = 0; nt < 8; ++nt) {
#pragma unroll
            for (int i = 0; i < 8; ++i) {
                long long row = tile * 16 + i + 8 * half;
                out[row * HID + nt * 16 + m] = acc[nt][i];
            }
        }
    }
}

// ---------------------------------------------------------------- output projection (HID -> 3)
__global__ __launch_bounds__(256) void scdm_out_lin(
    const float* __restrict__ x, const float* __restrict__ W_out,
    const float* __restrict__ b_out, float* __restrict__ out) {
    long long n = (long long)blockIdx.x * 256 + threadIdx.x;
    if (n >= N_NODES) return;
    float a0 = b_out[0], a1 = b_out[1], a2 = b_out[2];
    const float4* rp = (const float4*)(x + n * HID);
#pragma unroll 4
    for (int k4 = 0; k4 < HID / 4; ++k4) {
        float4 v = rp[k4];
        int k = k4 * 4;
        a0 += v.x * W_out[(k + 0) * OUTD + 0];
        a1 += v.x * W_out[(k + 0) * OUTD + 1];
        a2 += v.x * W_out[(k + 0) * OUTD + 2];
        a0 += v.y * W_out[(k + 1) * OUTD + 0];
        a1 += v.y * W_out[(k + 1) * OUTD + 1];
        a2 += v.y * W_out[(k + 1) * OUTD + 2];
        a0 += v.z * W_out[(k + 2) * OUTD + 0];
        a1 += v.z * W_out[(k + 2) * OUTD + 1];
        a2 += v.z * W_out[(k + 2) * OUTD + 2];
        a0 += v.w * W_out[(k + 3) * OUTD + 0];
        a1 += v.w * W_out[(k + 3) * OUTD + 1];
        a2 += v.w * W_out[(k + 3) * OUTD + 2];
    }
    out[n * OUTD + 0] = a0;
    out[n * OUTD + 1] = a1;
    out[n * OUTD + 2] = a2;
}

extern "C" void kernel_launch(void* const* d_in, const int* in_sizes, int n_in,
                              void* d_out, int out_size, void* d_ws, size_t ws_size,
                              hipStream_t stream) {
    const float* feature = (const float*)d_in[0];
    const int*   index   = (const int*)d_in[1];
    const int*   etype   = (const int*)d_in[2];
    const float* W_in    = (const float*)d_in[3];
    const float* b_in    = (const float*)d_in[4];
    const float* W_rel   = (const float*)d_in[5];
    const float* W_root  = (const float*)d_in[6];
    const float* b_rgcn  = (const float*)d_in[7];
    const float* W_out   = (const float*)d_in[8];
    const float* b_out   = (const float*)d_in[9];
    float* out = (float*)d_out;

    char* ws = (char*)d_ws;
    // scratch layout (bytes)
    float* agg  = (float*)(ws);                                   // 2*N*HID f32 = 102,400,000
    float* cnt  = (float*)(ws + 102400000LL);                     //   2*N f32  =     800,000
    float* xA   = (float*)(ws + 103200000LL);                     //   N*HID    =  51,200,000
    float* xB   = (float*)(ws + 154400000LL);                     //   N*HID    =  51,200,000
    unsigned short* wRoot = (unsigned short*)(ws + 205600000LL);  // 32,768
    unsigned short* wRel  = (unsigned short*)(ws + 205632768LL);  // 65,536
    unsigned short* wIn   = (unsigned short*)(ws + 205698304LL);  //  8,192 (32x128, zero-padded)

    const int tile_blocks = (NTILES + WAVES_PER_BLK - 1) / WAVES_PER_BLK;   // 782
    const long long agg_cnt_elems = 2LL * N_NODES * HID + 2LL * N_NODES;    // agg + cnt contiguous
    const long long agg_elems     = 2LL * N_NODES * HID;

    scdm_prep_weights<<<(2 * HID * HID + 255) / 256, 256, 0, stream>>>(
        W_in, W_root, W_rel, wIn, wRoot, wRel);

    scdm_zero<<<2048, 256, 0, stream>>>(agg, agg_cnt_elems);
    scdm_count<<<N_EDGES / 256, 256, 0, stream>>>(index, etype, cnt);

    scdm_mlp_in<<<tile_blocks, 256, 0, stream>>>(feature, wIn, b_in, xA);

    // layer 1
    scdm_scatter<<<(N_EDGES * 32) / 256, 256, 0, stream>>>(index, etype, xA, agg);
    scdm_rgcn_node<<<tile_blocks, 256, 0, stream>>>(xA, agg, cnt, wRoot, wRel, b_rgcn, xB);

    // layer 2 (same weights, same edges; cnt unchanged, agg re-zeroed)
    scdm_zero<<<2048, 256, 0, stream>>>(agg, agg_elems);
    scdm_scatter<<<(N_EDGES * 32) / 256, 256, 0, stream>>>(index, etype, xB, agg);
    scdm_rgcn_node<<<tile_blocks, 256, 0, stream>>>(xB, agg, cnt, wRoot, wRel, b_rgcn, xA);

    scdm_out_lin<<<(N_NODES + 255) / 256, 256, 0, stream>>>(xA, W_out, b_out, out);
}
